// HybridQLSTM_65481071404064
// MI455X (gfx1250) — matrix operand order repaired and checked by reading, not compiled, not run on hardware
//
#include <hip/hip_runtime.h>
#include <hip/hip_bf16.h>

#define S_LEN 512
#define B_SZ  64
#define DIN   256
#define HID   512
#define NH    4
#define DK    128
#define DCOMB 768   // DIN + HID
#define NG    2048  // 4 * HID

typedef __bf16 v16bf __attribute__((ext_vector_type(16)));
typedef float  v8f   __attribute__((ext_vector_type(8)));
typedef unsigned int v4u __attribute__((ext_vector_type(4)));

union Frag { v16bf v; v4u q[2]; };

__device__ inline unsigned short f2bf(float f) {
  unsigned int u = __float_as_uint(f);
  u += 0x7fffu + ((u >> 16) & 1u);          // round-to-nearest-even
  return (unsigned short)(u >> 16);
}

// Load a 16-half WMMA fragment as two 16-byte vectors.
// p0 -> halves [0..7], p1 -> halves [8..15] of the lane's fragment.
__device__ inline v16bf ld_frag(const unsigned short* p0, const unsigned short* p1) {
  Frag f;
  f.q[0] = *reinterpret_cast<const v4u*>(p0);
  f.q[1] = *reinterpret_cast<const v4u*>(p1);
  return f.v;
}

__device__ inline v8f wmma_bf16(v16bf a, v16bf b, v8f c) {
  // D = A(16x32 bf16) * B(32x16 bf16) + C(16x16 f32)
  return __builtin_amdgcn_wmma_f32_16x16x32_bf16(
      false, a, false, b, (short)0, c, false, false);
}

__device__ inline float sigmoidf_(float x) { return 1.f / (1.f + __expf(-x)); }

// Device-wide barrier for the persistent LSTM kernel (32 co-resident blocks).
// Release fence + block barrier; only thread 0 arrives and spins with RELAXED
// polls (no per-poll cache invalidate); after release, one acquire fence per
// thread/wave invalidates the WGP$ before h is re-read.
__device__ inline void grid_barrier(unsigned int* bar, unsigned int target) {
  __threadfence();                 // release: flush this wave's h stores
  __syncthreads();
  if (threadIdx.x == 0) {
    __hip_atomic_fetch_add(bar, 1u, __ATOMIC_RELAXED, __HIP_MEMORY_SCOPE_AGENT);
    while (__hip_atomic_load(bar, __ATOMIC_RELAXED, __HIP_MEMORY_SCOPE_AGENT) < target)
      __builtin_amdgcn_s_sleep(2);
  }
  __syncthreads();
  __threadfence();                 // acquire: invalidate L0 before reading h
}

// ---------------------------------------------------------------- prep kernels

__global__ void cvt_bf16(const float* __restrict__ src,
                         unsigned short* __restrict__ dst, int n) {
  int i = blockIdx.x * blockDim.x + threadIdx.x;
  if (i < n) dst[i] = f2bf(src[i]);
}

// Build WlstmT[2048][768] (bf16, row = gate output unit, col = fan-in) and bias[2048].
__global__ void prep_wlstm(const float* __restrict__ Wf, const float* __restrict__ Wi,
                           const float* __restrict__ Wg, const float* __restrict__ Wo,
                           const float* __restrict__ bf_, const float* __restrict__ bi_,
                           const float* __restrict__ bg_, const float* __restrict__ bo_,
                           unsigned short* __restrict__ WT, float* __restrict__ bias) {
  int idx = blockIdx.x * blockDim.x + threadIdx.x;
  if (idx >= NG * DCOMB) return;
  int n = idx / DCOMB, k = idx - n * DCOMB;
  int g = n >> 9, j = n & 511;
  const float* W = (g == 0) ? Wf : (g == 1) ? Wi : (g == 2) ? Wg : Wo;
  WT[idx] = f2bf(W[k * HID + j]);   // W stored (in, out): W[k][j]
  if (k == 0) {
    const float* bb = (g == 0) ? bf_ : (g == 1) ? bi_ : (g == 2) ? bg_ : bo_;
    bias[n] = bb[j];
  }
}

// WT[n][k] = W[k][n] for 512x512 projection weights.
__global__ void transpose512(const float* __restrict__ W,
                             unsigned short* __restrict__ WT) {
  int idx = blockIdx.x * blockDim.x + threadIdx.x;
  int n = idx >> 9, k = idx & 511;
  WT[idx] = f2bf(W[k * HID + n]);
}

__global__ void zero_bar(unsigned int* __restrict__ bar) {
  if (threadIdx.x == 0 && blockIdx.x == 0) *bar = 0u;
}

// ---------------------------------------------------------------- persistent LSTM

// One launch for all 512 timesteps. 32 blocks x 128 threads = 128 waves
// (1 wave/SIMD so the ~1000-VGPR weight-resident waves always fit).
// Wave (mt, jt) owns batch rows mt*16..+15 and hidden cols jt*16..+15:
//   - computes all 4 gate tiles (A fragment reused 4x per K-chunk, 96 WMMA/step)
//   - weight fragments are loop-invariant -> compiler keeps them in the
//     extended VGPR file (s_set_vgpr_msb), loaded once for all 512 steps
//   - cell state c lives in VGPRs for the whole sequence
//   - writes h (bf16) to hbf + lstm_out, then grid-barriers before next step.
__global__ __launch_bounds__(128) void lstm_persist(
    const unsigned short* __restrict__ xbf,     // [S][64][256]
    unsigned short* __restrict__ hbf,           // [64][512]
    const unsigned short* __restrict__ WT,      // [2048][768]
    const float* __restrict__ bias,             // [2048]
    unsigned short* __restrict__ lobf,          // [S*64][512]
    float* __restrict__ hx, float* __restrict__ cx,
    unsigned int* __restrict__ bar) {
  int lane = threadIdx.x & 31, wid = threadIdx.x >> 5;
  int wg = blockIdx.x * 4 + wid;        // 0..127
  int mt = wg & 3, jt = wg >> 2;        // 4 batch-tiles x 32 hidden-tiles
  int hi = lane >> 4, ln = lane & 15;
  int ak0 = hi ? 16 : 0;                // A-frag half-select (contiguous k = 16*hi + s)
  int bk0 = hi ? 16 : 0;                // B-frag K-half select
  int arow = mt * 16 + ln;
  int col  = jt * 16 + ln;

  const unsigned short* bp0 = WT + (size_t)(0 * HID + jt * 16 + ln) * DCOMB;
  const unsigned short* bp1 = WT + (size_t)(1 * HID + jt * 16 + ln) * DCOMB;
  const unsigned short* bp2 = WT + (size_t)(2 * HID + jt * 16 + ln) * DCOMB;
  const unsigned short* bp3 = WT + (size_t)(3 * HID + jt * 16 + ln) * DCOMB;
  float bF = bias[col], bI = bias[HID + col], bG = bias[2 * HID + col],
        bO = bias[3 * HID + col];

  // init: zero my h patch, c lives in registers
  float creg[8];
#pragma unroll
  for (int i = 0; i < 8; ++i) {
    creg[i] = 0.f;
    hbf[(mt * 16 + i + hi * 8) * HID + col] = 0;
  }
  unsigned int tgt = 32;
  grid_barrier(bar, tgt);

  for (int t = 0; t < S_LEN; ++t) {
    v8f aF = {}, aI = {}, aG = {}, aO = {};
    const unsigned short* xt = xbf + (size_t)t * B_SZ * DIN;
#pragma unroll
    for (int kc = 0; kc < 24; ++kc) {
      const unsigned short* ab = (kc < 8)
          ? (xt + arow * DIN + kc * 32)
          : (hbf + arow * HID + (kc - 8) * 32);
      v16bf a = ld_frag(ab + ak0, ab + ak0 + 8);
      v16bf b0 = ld_frag(bp0 + kc * 32 + bk0, bp0 + kc * 32 + bk0 + 8);
      aF = wmma_bf16(a, b0, aF);
      v16bf b1 = ld_frag(bp1 + kc * 32 + bk0, bp1 + kc * 32 + bk0 + 8);
      aI = wmma_bf16(a, b1, aI);
      v16bf b2 = ld_frag(bp2 + kc * 32 + bk0, bp2 + kc * 32 + bk0 + 8);
      aG = wmma_bf16(a, b2, aG);
      v16bf b3 = ld_frag(bp3 + kc * 32 + bk0, bp3 + kc * 32 + bk0 + 8);
      aO = wmma_bf16(a, b3, aO);
    }
#pragma unroll
    for (int i = 0; i < 8; ++i) {
      float f  = sigmoidf_(aF[i] + bF);
      float ii = sigmoidf_(aI[i] + bI);
      float gg = tanhf(aG[i] + bG);
      float o  = sigmoidf_(aO[i] + bO);
      float c  = f * creg[i] + ii * gg;
      float h  = o * tanhf(c);
      creg[i] = c;
      int row = mt * 16 + i + hi * 8;         // C-layout: vgpr i -> rows i / i+8
      unsigned short hb = f2bf(h);
      hbf[row * HID + col] = hb;
      lobf[((size_t)t * B_SZ + row) * HID + col] = hb;
      if (t == S_LEN - 1) {
        hx[row * HID + col] = h;
        cx[row * HID + col] = c;
      }
    }
    tgt += 32;
    grid_barrier(bar, tgt);                   // h visible before next step
  }
}

// ---------------------------------------------------------------- projections

// P = lstm_out_bf[32768][512] @ W + b, stored bf16 per layout:
//   vmode==0: out[((b*NH+h)*S + s)*DK + d]   (q, k: rows = tokens, contiguous dk)
//   vmode==1: out[((b*NH+h)*DK + d)*S + s]   (v transposed: contiguous keys)
// where r=b*S+s (faithful to reference's flat reinterpretation), c=h*DK+d.
__global__ __launch_bounds__(256) void proj_gemm(
    const unsigned short* __restrict__ A,    // [32768][512]
    const unsigned short* __restrict__ WT,   // [512][512]
    const float* __restrict__ bias,
    unsigned short* __restrict__ out, int vmode) {
  int lane = threadIdx.x & 31, wid = threadIdx.x >> 5;
  int wg = blockIdx.x * 8 + wid;        // 0..65535
  int nt = wg & 31, mt = wg >> 5;       // 32 N-tiles x 2048 M-tiles
  int hi = lane >> 4, ln = lane & 15;
  int ak0 = hi ? 16 : 0, bk0 = hi ? 16 : 0;
  const unsigned short* arow = A + (size_t)(mt * 16 + ln) * HID;
  const unsigned short* brow = WT + (size_t)(nt * 16 + ln) * HID;
  v8f acc = {};
#pragma unroll
  for (int kc = 0; kc < 16; ++kc) {
    v16bf a = ld_frag(arow + kc * 32 + ak0, arow + kc * 32 + ak0 + 8);
    v16bf b = ld_frag(brow + kc * 32 + bk0, brow + kc * 32 + bk0 + 8);
    acc = wmma_bf16(a, b, acc);
  }
  int c = nt * 16 + ln;
  float bv = bias[c];
  int h = c >> 7, d = c & 127;
#pragma unroll
  for (int i = 0; i < 8; ++i) {
    int r = mt * 16 + i + hi * 8;
    int bb = r >> 9, s = r & 511;
    unsigned short ov = f2bf(acc[i] + bv);
    if (vmode) out[((bb * NH + h) * DK + d) * S_LEN + s] = ov;
    else       out[((bb * NH + h) * S_LEN + s) * DK + d] = ov;
  }
}

// ---------------------------------------------------------------- flash attention

// One wave per (b,h, 16-query tile). Streams 16 chunks of 32 keys with online
// softmax; P tile goes C-layout(f32) -> LDS(bf16) -> A-layout for the P@V WMMA.
__global__ __launch_bounds__(256) void attn_kernel(
    const unsigned short* __restrict__ Q,    // [bh][s][dk]
    const unsigned short* __restrict__ K,    // [bh][s][dk]
    const unsigned short* __restrict__ Vt,   // [bh][dk][s]
    float* __restrict__ decoded) {           // [S][B][HID]
  __shared__ __align__(16) unsigned short ptile[8 * 16 * 32];
  int lane = threadIdx.x & 31, wid = threadIdx.x >> 5;
  int wg = blockIdx.x * 8 + wid;        // 0..8191
  int qt = wg & 31, bh = wg >> 5;       // 32 q-tiles x 256 (b,h)
  int hi = lane >> 4, ln = lane & 15;
  int ak0 = hi ? 16 : 0, bk0 = hi ? 16 : 0;

  const unsigned short* qbase = Q + (size_t)(bh * S_LEN + qt * 16 + ln) * DK;
  v16bf qf[4];
#pragma unroll
  for (int kc = 0; kc < 4; ++kc)
    qf[kc] = ld_frag(qbase + kc * 32 + ak0, qbase + kc * 32 + ak0 + 8);

  v8f zero = {};
  v8f acc[8];
  float mrow[8], lrow[8];
#pragma unroll
  for (int dt = 0; dt < 8; ++dt) acc[dt] = zero;
#pragma unroll
  for (int i = 0; i < 8; ++i) { mrow[i] = -3.4e38f; lrow[i] = 0.f; }

  unsigned short* myp = ptile + wid * 16 * 32;
  const float scale = 0.08838834764831845f;   // 1/sqrt(128)

  for (int kb = 0; kb < 16; ++kb) {
    v8f s0 = zero, s1 = zero;
    const unsigned short* kr0 = K + (size_t)(bh * S_LEN + kb * 32 + ln) * DK;
    const unsigned short* kr1 = K + (size_t)(bh * S_LEN + kb * 32 + 16 + ln) * DK;
#pragma unroll
    for (int kc = 0; kc < 4; ++kc) {
      v16bf b0 = ld_frag(kr0 + kc * 32 + bk0, kr0 + kc * 32 + bk0 + 8);
      s0 = wmma_bf16(qf[kc], b0, s0);
      v16bf b1 = ld_frag(kr1 + kc * 32 + bk0, kr1 + kc * 32 + bk0 + 8);
      s1 = wmma_bf16(qf[kc], b1, s1);
    }
#pragma unroll
    for (int i = 0; i < 8; ++i) {
      float a = s0[i] * scale, b = s1[i] * scale;
      // row max across the 16 lanes holding this row (xor masks stay in-group)
      float t = fmaxf(a, b);
      t = fmaxf(t, __shfl_xor(t, 8));
      t = fmaxf(t, __shfl_xor(t, 4));
      t = fmaxf(t, __shfl_xor(t, 2));
      t = fmaxf(t, __shfl_xor(t, 1));
      float mn = fmaxf(mrow[i], t);
      float alpha = __expf(mrow[i] - mn);
      float p0 = __expf(a - mn);
      float p1 = __expf(b - mn);
      float rs = p0 + p1;
      rs += __shfl_xor(rs, 8);
      rs += __shfl_xor(rs, 4);
      rs += __shfl_xor(rs, 2);
      rs += __shfl_xor(rs, 1);
      lrow[i] = lrow[i] * alpha + rs;
      mrow[i] = mn;
#pragma unroll
      for (int dt = 0; dt < 8; ++dt) acc[dt][i] *= alpha;
      int prow = i + hi * 8;                       // C-layout row
      myp[prow * 32 + ln]      = f2bf(p0);
      myp[prow * 32 + 16 + ln] = f2bf(p1);
    }
    // Reload P in A-fragment layout (wave-local LDS, in-order DS pipe).
    v16bf pf = ld_frag(myp + ln * 32 + ak0, myp + ln * 32 + ak0 + 8);
#pragma unroll
    for (int dt = 0; dt < 8; ++dt) {
      const unsigned short* vr =
          Vt + (size_t)(bh * DK + dt * 16 + ln) * S_LEN + kb * 32 + bk0;
      v16bf bv = ld_frag(vr, vr + 8);
      acc[dt] = wmma_bf16(pf, bv, acc[dt]);
    }
  }
  int b = bh >> 2, h = bh & 3;
#pragma unroll
  for (int i = 0; i < 8; ++i) {
    float inv = 1.f / lrow[i];
    int s = qt * 16 + i + hi * 8;
#pragma unroll
    for (int dt = 0; dt < 8; ++dt) {
      int col = h * DK + dt * 16 + ln;
      decoded[((size_t)s * B_SZ + b) * HID + col] = acc[dt][i] * inv;
    }
  }
}

// ---------------------------------------------------------------- launch

extern "C" void kernel_launch(void* const* d_in, const int* in_sizes, int n_in,
                              void* d_out, int out_size, void* d_ws, size_t ws_size,
                              hipStream_t stream) {
  const float* x  = (const float*)d_in[0];
  const float* Wf = (const float*)d_in[1];  const float* bf_ = (const float*)d_in[2];
  const float* Wi = (const float*)d_in[3];  const float* bi_ = (const float*)d_in[4];
  const float* Wg = (const float*)d_in[5];  const float* bg_ = (const float*)d_in[6];
  const float* Wo = (const float*)d_in[7];  const float* bo_ = (const float*)d_in[8];
  const float* Wq = (const float*)d_in[9];  const float* bq  = (const float*)d_in[10];
  const float* Wk = (const float*)d_in[11]; const float* bk  = (const float*)d_in[12];
  const float* Wv = (const float*)d_in[13]; const float* bv  = (const float*)d_in[14];

  float* decoded = (float*)d_out;                       // [S][B][HID]
  float* hx = decoded + (size_t)S_LEN * B_SZ * HID;     // [B][HID]
  float* cx = hx + (size_t)B_SZ * HID;                  // [B][HID]

  unsigned char* ws = (unsigned char*)d_ws;
  size_t off = 0;
  auto take = [&](size_t bytes) {
    size_t o = off; off = (off + bytes + 255) & ~(size_t)255; return o;
  };
  unsigned short* xbf    = (unsigned short*)(ws + take((size_t)S_LEN * B_SZ * DIN * 2));
  unsigned short* WlstmT = (unsigned short*)(ws + take((size_t)NG * DCOMB * 2));
  float*          blstm  = (float*)(ws + take((size_t)NG * 4));
  unsigned short* hbf    = (unsigned short*)(ws + take((size_t)B_SZ * HID * 2));
  unsigned int*   bar    = (unsigned int*)(ws + take(256));
  unsigned short* lobf   = (unsigned short*)(ws + take((size_t)S_LEN * B_SZ * HID * 2));
  unsigned short* WqT    = (unsigned short*)(ws + take((size_t)HID * HID * 2));
  unsigned short* WkT    = (unsigned short*)(ws + take((size_t)HID * HID * 2));
  unsigned short* WvT    = (unsigned short*)(ws + take((size_t)HID * HID * 2));
  unsigned short* qbf    = (unsigned short*)(ws + take((size_t)S_LEN * B_SZ * HID * 2));
  unsigned short* kbf    = (unsigned short*)(ws + take((size_t)S_LEN * B_SZ * HID * 2));
  unsigned short* vbfT   = (unsigned short*)(ws + take((size_t)S_LEN * B_SZ * HID * 2));
  (void)ws_size;

  // prep
  int nx = S_LEN * B_SZ * DIN;
  cvt_bf16<<<(nx + 255) / 256, 256, 0, stream>>>(x, xbf, nx);
  prep_wlstm<<<(NG * DCOMB + 255) / 256, 256, 0, stream>>>(
      Wf, Wi, Wg, Wo, bf_, bi_, bg_, bo_, WlstmT, blstm);
  transpose512<<<(HID * HID) / 256, 256, 0, stream>>>(Wq, WqT);
  transpose512<<<(HID * HID) / 256, 256, 0, stream>>>(Wk, WkT);
  transpose512<<<(HID * HID) / 256, 256, 0, stream>>>(Wv, WvT);
  zero_bar<<<1, 32, 0, stream>>>(bar);

  // whole sequential LSTM in ONE persistent launch (32 co-resident blocks)
  lstm_persist<<<32, 128, 0, stream>>>(xbf, hbf, WlstmT, blstm, lobf, hx, cx, bar);

  // q/k/v projections (65536 tiles, 8 waves per block)
  proj_gemm<<<8192, 256, 0, stream>>>(lobf, WqT, bq, qbf, 0);
  proj_gemm<<<8192, 256, 0, stream>>>(lobf, WkT, bk, kbf, 0);
  proj_gemm<<<8192, 256, 0, stream>>>(lobf, WvT, bv, vbfT, 1);

  // fused flash attention (8192 waves)
  attn_kernel<<<1024, 256, 0, stream>>>(qbf, kbf, vbfT, decoded);
}
